// MultiHeadSelfAttention_4896262717658
// MI455X (gfx1250) — matrix operand order repaired and checked by reading, not compiled, not run on hardware
//
#include <hip/hip_runtime.h>
#include <hip/hip_bf16.h>

// MI455X (gfx1250) fused multi-head self-attention, wave32 + WMMA +
// double-buffered ASYNC global->LDS tile staging.
// Pipeline: [cvt weights f16] -> [LayerNorm->f16] -> [WMMA GEMM Q,K,Vt] ->
//           [flash attention, WMMA QK^T + online softmax + WMMA PV] ->
//           [WMMA GEMM fc -> f32 out]

typedef __attribute__((ext_vector_type(16))) _Float16 v16h;
typedef __attribute__((ext_vector_type(8)))  _Float16 v8h;
typedef __attribute__((ext_vector_type(8)))  float    v8f;
typedef int a16i __attribute__((vector_size(16)));   // int4 for async-LDS builtin

#define BB   2
#define SS   4096
#define DIM  512
#define NH   8
#define HD   64
#define ROWS (BB*SS)          // 8192
#define ATT_SCALE 0.125f      // 64^-0.5
#define LN_EPS 1e-5f

#ifndef __has_builtin
#define __has_builtin(x) 0
#endif
#if __has_builtin(__builtin_amdgcn_global_load_async_to_lds_b128)
#define ASYNC_LDS 1
#else
#define ASYNC_LDS 0
#endif

static __device__ __forceinline__ v8h ld8(const _Float16* p) {
    return *(const v8h*)p;
}
static __device__ __forceinline__ v16h mk16(v8h a, v8h b) {
    return __builtin_shufflevector(a, b, 0,1,2,3,4,5,6,7,8,9,10,11,12,13,14,15);
}
static __device__ __forceinline__ v8f wmma32(v16h a, v16h b, v8f c) {
    // D = A(16x32 f16) x B(32x16 f16) + C(16x16 f32)
    return __builtin_amdgcn_wmma_f32_16x16x32_f16(false, a, false, b, (short)0, c, false, false);
}

// 16-byte global -> LDS copy. Async (ASYNCcnt) when available, else direct.
static __device__ __forceinline__ void async_cp16(const _Float16* g, _Float16* l) {
#if ASYNC_LDS
    __builtin_amdgcn_global_load_async_to_lds_b128(
        (__attribute__((address_space(1))) a16i*)(unsigned long long)g,
        (__attribute__((address_space(3))) a16i*)(unsigned int)(unsigned long long)l,
        0, 0);
#else
    *(v8h*)l = *(const v8h*)g;
#endif
}

template <int N>
static __device__ __forceinline__ void wait_async() {
#if ASYNC_LDS
#if __has_builtin(__builtin_amdgcn_s_wait_asynccnt)
    __builtin_amdgcn_s_wait_asynccnt(N);
#else
    asm volatile("s_wait_asynccnt %0" :: "i"(N) : "memory");
#endif
#endif
}

// ---------------------------------------------------------------------------
// Kernel 1: convert the four 512x512 f32 weights to f16.
// ---------------------------------------------------------------------------
__global__ __launch_bounds__(256) void mha_cvt_w(
    const float* __restrict__ wq, const float* __restrict__ wk,
    const float* __restrict__ wv, const float* __restrict__ wf,
    _Float16* __restrict__ oq, _Float16* __restrict__ ok,
    _Float16* __restrict__ ov, _Float16* __restrict__ of) {
    int i = blockIdx.x * 256 + threadIdx.x;   // grid covers 512*512 exactly
    oq[i] = (_Float16)wq[i];
    ok[i] = (_Float16)wk[i];
    ov[i] = (_Float16)wv[i];
    of[i] = (_Float16)wf[i];
}

// ---------------------------------------------------------------------------
// Kernel 2: LayerNorm over DIM=512, output f16. One wave32 per row.
// ---------------------------------------------------------------------------
__global__ __launch_bounds__(256) void mha_ln(
    const float* __restrict__ x, const float* __restrict__ gamma,
    const float* __restrict__ beta, _Float16* __restrict__ xn) {
    const int row  = blockIdx.x * 8 + (threadIdx.x >> 5);
    const int lane = threadIdx.x & 31;
    const float* xr = x + (size_t)row * DIM;
    float v[16], s = 0.f, ss = 0.f;
#pragma unroll
    for (int i = 0; i < 16; ++i) {
        float f = xr[lane + 32 * i];
        v[i] = f; s += f; ss += f * f;
    }
#pragma unroll
    for (int m = 1; m < 32; m <<= 1) {
        s  += __shfl_xor(s,  m, 32);
        ss += __shfl_xor(ss, m, 32);
    }
    const float mu  = s * (1.f / DIM);
    const float var = ss * (1.f / DIM) - mu * mu;
    const float rs  = rsqrtf(var + LN_EPS);
    _Float16* orow = xn + (size_t)row * DIM;
#pragma unroll
    for (int i = 0; i < 16; ++i) {
        int c = lane + 32 * i;
        orow[c] = (_Float16)((v[i] - mu) * rs * gamma[c] + beta[c]);
    }
}

// ---------------------------------------------------------------------------
// Kernel 3: GEMM  C[M,N] = A[M,512] * W[N,512]^T  (f16 in, f32 accum).
//   mode 0: C16 row-major f16            (Q, K)
//   mode 1: C16 scattered as Vt[b,h,d,s] (V transposed per head)
//   mode 2: C32 row-major f32            (final fc output)
// Block: 256 thr = 8 waves; tile 128(M) x 64(N); K-chunk 64; double-buffered
// async global->LDS staging.
// ---------------------------------------------------------------------------
__global__ __launch_bounds__(256) void mha_gemm(
    const _Float16* __restrict__ A, const _Float16* __restrict__ W,
    _Float16* __restrict__ C16, float* __restrict__ C32, int mode) {
    __shared__ alignas(16) _Float16 At[2][128][72];
    __shared__ alignas(16) _Float16 Bt[2][64][72];

    const int t = threadIdx.x, w = t >> 5, lane = t & 31;
    const int nl = lane & 15, hi16 = (lane >= 16) ? 1 : 0;
    const int mblk = blockIdx.x * 128, nblk = blockIdx.y * 64;

    const int ar = t >> 1, asg = (t & 1) * 32;   // A staging: 64B / thread
    const int br = t >> 2, bsg = (t & 3) * 16;   // B staging: 32B / thread

    // 6 async 16B copies per thread per K-chunk.
    auto stage = [&](int k0, int buf) {
        const _Float16* sa = A + (size_t)(mblk + ar) * 512 + k0 + asg;
        _Float16* da = &At[buf][ar][asg];
#pragma unroll
        for (int j = 0; j < 4; ++j) async_cp16(sa + j * 8, da + j * 8);
        const _Float16* sb = W + (size_t)(nblk + br) * 512 + k0 + bsg;
        _Float16* db = &Bt[buf][br][bsg];
#pragma unroll
        for (int j = 0; j < 2; ++j) async_cp16(sb + j * 8, db + j * 8);
    };

    const v8f vzero = {0.f,0.f,0.f,0.f,0.f,0.f,0.f,0.f};
    v8f acc[4] = {vzero, vzero, vzero, vzero};

    stage(0, 0);
    for (int i = 0; i < 8; ++i) {               // K = 512 = 8 chunks of 64
        const int buf = i & 1;
        if (i < 7) { stage((i + 1) * 64, buf ^ 1); wait_async<6>(); }
        else       { wait_async<0>(); }
        __syncthreads();                        // tile i visible to all waves
#pragma unroll
        for (int kk = 0; kk < 2; ++kk) {
            const int aoff = kk * 32 + (hi16 ? 8 : 0);     // ISA A-fragment layout
            v16h a = mk16(ld8(&At[buf][w * 16 + nl][aoff]),
                          ld8(&At[buf][w * 16 + nl][aoff + 16]));
            const int boff = kk * 32 + (hi16 ? 8 : 0);     // ISA B-fragment layout
#pragma unroll
            for (int nt = 0; nt < 4; ++nt) {
                v16h b = mk16(ld8(&Bt[buf][nt * 16 + nl][boff]),
                              ld8(&Bt[buf][nt * 16 + nl][boff + 16]));
                acc[nt] = wmma32(a, b, acc[nt]);
            }
        }
        __syncthreads();                        // done reading buf before restage
    }

    const int mtop = hi16 * 8;
    if (mode == 1) {
        // Vt[b][h][d][s]; the 8 C rows are 8 consecutive s -> one v8h store.
        const int sg0 = mblk + w * 16 + mtop;
        const int b = sg0 >> 12, s = sg0 & (SS - 1);
#pragma unroll
        for (int nt = 0; nt < 4; ++nt) {
            const int e = nblk + nt * 16 + nl;
            const int h = e >> 6, d = e & 63;
            v8h hv;
#pragma unroll
            for (int r = 0; r < 8; ++r) hv[r] = (_Float16)acc[nt][r];
            *(v8h*)(C16 + (((size_t)(b * NH + h) * HD + d) * SS + s)) = hv;
        }
    } else if (mode == 0) {
#pragma unroll
        for (int nt = 0; nt < 4; ++nt)
#pragma unroll
            for (int r = 0; r < 8; ++r)
                C16[(size_t)(mblk + w * 16 + mtop + r) * 512 + nblk + nt * 16 + nl] =
                    (_Float16)acc[nt][r];
    } else {
#pragma unroll
        for (int nt = 0; nt < 4; ++nt)
#pragma unroll
            for (int r = 0; r < 8; ++r)
                C32[(size_t)(mblk + w * 16 + mtop + r) * 512 + nblk + nt * 16 + nl] =
                    acc[nt][r];
    }
}

// ---------------------------------------------------------------------------
// Kernel 4: flash attention. grid = (S/128, B*H), block = 256 (8 waves).
// Each wave owns 16 query rows; streams 64-wide K/V tiles through LDS with
// double-buffered async staging.
// ---------------------------------------------------------------------------
__global__ __launch_bounds__(256) void mha_attn(
    const _Float16* __restrict__ Q, const _Float16* __restrict__ K,
    const _Float16* __restrict__ Vt, _Float16* __restrict__ O) {
    __shared__ alignas(16) _Float16 Kt [2][64][72];  // K rows (kcol) x d
    __shared__ alignas(16) _Float16 Vts[2][64][72];  // Vt rows (d)   x kcol
    __shared__ float              Ssc[8][16][66];    // per-wave f32 scores
    __shared__ alignas(16) _Float16 Pt [8][16][72];  // per-wave f16 probs
    __shared__ float              Ab [8][16];        // per-wave row scale

    const int t = threadIdx.x, w = t >> 5, lane = t & 31;
    const int nl = lane & 15, hi16 = (lane >= 16) ? 1 : 0;
    const int mtop = hi16 * 8;
    const int qblk = blockIdx.x * 128;
    const int bh = blockIdx.y, b = bh >> 3, h = bh & 7;

    // Q fragments for this wave's 16 rows (reduction dim d = 64 -> 2 chunks).
    const int qrow = qblk + w * 16 + nl;
    const _Float16* Qp = Q + ((size_t)(b * SS + qrow)) * DIM + h * HD;
    v16h aQ[2];
#pragma unroll
    for (int kk = 0; kk < 2; ++kk) {
        const int off = kk * 32 + (hi16 ? 8 : 0);
        aQ[kk] = mk16(ld8(Qp + off), ld8(Qp + off + 16));
    }

    const v8f vzero = {0.f,0.f,0.f,0.f,0.f,0.f,0.f,0.f};
    v8f accO[4] = {vzero, vzero, vzero, vzero};
    float mrow = -3.4e38f, lrow = 0.f;

    const _Float16* Kbase = K  + ((size_t)(b * SS)) * DIM + h * HD;
    const _Float16* Vbase = Vt + ((size_t)bh) * HD * SS;
    const int ldr = t >> 2, ldsg = (t & 3) * 16;

    // 4 async 16B copies per thread per 64-wide K/V tile.
    auto stageKV = [&](int kt0, int buf) {
        const _Float16* sk = Kbase + (size_t)(kt0 + ldr) * DIM + ldsg;
        async_cp16(sk,     &Kt[buf][ldr][ldsg]);
        async_cp16(sk + 8, &Kt[buf][ldr][ldsg + 8]);
        const _Float16* sv = Vbase + (size_t)ldr * SS + kt0 + ldsg;
        async_cp16(sv,     &Vts[buf][ldr][ldsg]);
        async_cp16(sv + 8, &Vts[buf][ldr][ldsg + 8]);
    };

    stageKV(0, 0);
    for (int it = 0; it < SS / 64; ++it) {
        const int buf = it & 1;
        if (it < SS / 64 - 1) { stageKV((it + 1) * 64, buf ^ 1); wait_async<4>(); }
        else                  { wait_async<0>(); }
        __syncthreads();                 // tile `it` visible to all waves

        // S = scale * Q K^T  (4 column tiles of 16)
#pragma unroll
        for (int nt = 0; nt < 4; ++nt) {
            v8f c = vzero;
#pragma unroll
            for (int kk = 0; kk < 2; ++kk) {
                const int boff = kk * 32 + (hi16 ? 8 : 0);
                v16h bk = mk16(ld8(&Kt[buf][nt * 16 + nl][boff]),
                               ld8(&Kt[buf][nt * 16 + nl][boff + 16]));
                c = wmma32(aQ[kk], bk, c);
            }
#pragma unroll
            for (int r = 0; r < 8; ++r)
                Ssc[w][mtop + r][nt * 16 + nl] = c[r] * ATT_SCALE;
        }
        __syncthreads();

        // Online softmax: lane pair (nl, nl+16) handles row nl (32 cols each).
        {
            const float* srow = Ssc[w][nl];
            const int cb = hi16 * 32;
            float mloc = -3.4e38f;
#pragma unroll
            for (int j = 0; j < 32; ++j) mloc = fmaxf(mloc, srow[cb + j]);
            mloc = fmaxf(mloc, __shfl_xor(mloc, 16, 32));
            const float mnew  = fmaxf(mrow, mloc);
            const float alpha = __expf(mrow - mnew);
            float ps = 0.f;
#pragma unroll
            for (int j = 0; j < 32; ++j) {
                float p = __expf(srow[cb + j] - mnew);
                Pt[w][nl][cb + j] = (_Float16)p;
                ps += p;
            }
            ps += __shfl_xor(ps, 16, 32);
            lrow = lrow * alpha + ps;
            mrow = mnew;
            if (!hi16) Ab[w][nl] = alpha;
        }
        __syncthreads();

        // Rescale O accumulators, then O += P * V.
        float am[8];
#pragma unroll
        for (int r = 0; r < 8; ++r) am[r] = Ab[w][mtop + r];
#pragma unroll
        for (int nt = 0; nt < 4; ++nt)
#pragma unroll
            for (int r = 0; r < 8; ++r) accO[nt][r] *= am[r];

#pragma unroll
        for (int kk = 0; kk < 2; ++kk) {
            const int aoff = kk * 32 + (hi16 ? 8 : 0);
            v16h ap = mk16(ld8(&Pt[w][nl][aoff]), ld8(&Pt[w][nl][aoff + 16]));
            const int boff = kk * 32 + (hi16 ? 8 : 0);
#pragma unroll
            for (int nt = 0; nt < 4; ++nt) {
                v16h bv = mk16(ld8(&Vts[buf][nt * 16 + nl][boff]),
                               ld8(&Vts[buf][nt * 16 + nl][boff + 16]));
                accO[nt] = wmma32(ap, bv, accO[nt]);
            }
        }
        __syncthreads();                 // done reading buf before restage
    }

    // Epilogue: O /= l, write f16 [B,S,H,HD] (== [B,S,DIM]).
    if (!hi16) Ab[w][nl] = 1.0f / lrow;
    __syncthreads();
    float lm[8];
#pragma unroll
    for (int r = 0; r < 8; ++r) lm[r] = Ab[w][mtop + r];
#pragma unroll
    for (int nt = 0; nt < 4; ++nt)
#pragma unroll
        for (int r = 0; r < 8; ++r) {
            const int q = qblk + w * 16 + mtop + r;
            O[((size_t)(b * SS + q)) * DIM + h * HD + nt * 16 + nl] =
                (_Float16)(accO[nt][r] * lm[r]);
        }
}

// ---------------------------------------------------------------------------
extern "C" void kernel_launch(void* const* d_in, const int* in_sizes, int n_in,
                              void* d_out, int out_size, void* d_ws, size_t ws_size,
                              hipStream_t stream) {
    const float* x   = (const float*)d_in[0];
    const float* g   = (const float*)d_in[1];
    const float* be  = (const float*)d_in[2];
    const float* wq  = (const float*)d_in[3];
    const float* wk  = (const float*)d_in[4];
    const float* wv  = (const float*)d_in[5];
    const float* wf  = (const float*)d_in[6];
    float* out = (float*)d_out;

    char* ws = (char*)d_ws;
    const size_t MAT = (size_t)ROWS * DIM * sizeof(_Float16);   // 8 MB
    _Float16* xn   = (_Float16*)(ws);
    _Float16* q16  = (_Float16*)(ws + MAT);
    _Float16* k16  = (_Float16*)(ws + 2 * MAT);
    _Float16* vt16 = (_Float16*)(ws + 3 * MAT);
    _Float16* o16  = (_Float16*)(ws + 4 * MAT);
    _Float16* wq16 = (_Float16*)(ws + 5 * MAT);
    _Float16* wk16 = wq16 + (size_t)DIM * DIM;
    _Float16* wv16 = wk16 + (size_t)DIM * DIM;
    _Float16* wf16 = wv16 + (size_t)DIM * DIM;

    mha_cvt_w<<<dim3(DIM * DIM / 256), dim3(256), 0, stream>>>(
        wq, wk, wv, wf, wq16, wk16, wv16, wf16);
    mha_ln<<<dim3(ROWS / 8), dim3(256), 0, stream>>>(x, g, be, xn);

    dim3 gg(ROWS / 128, DIM / 64);
    mha_gemm<<<gg, dim3(256), 0, stream>>>(xn, wq16, q16,  nullptr, 0);
    mha_gemm<<<gg, dim3(256), 0, stream>>>(xn, wk16, k16,  nullptr, 0);
    mha_gemm<<<gg, dim3(256), 0, stream>>>(xn, wv16, vt16, nullptr, 1);

    mha_attn<<<dim3(SS / 128, BB * NH), dim3(256), 0, stream>>>(q16, k16, vt16, o16);

    mha_gemm<<<gg, dim3(256), 0, stream>>>(o16, wf16, nullptr, out, 2);
}
